// LinearAttention_50878182588744
// MI455X (gfx1250) — matrix operand order repaired and checked by reading, not compiled, not run on hardware
//
#include <hip/hip_runtime.h>

// ---------------------------------------------------------------------------
// Causal linear attention (ELU+1 feature map) for MI455X / gfx1250.
// B=2, T=2048, d_model=512, H=8, D=64.  Chunked (C=16) parallel-scan
// formulation: everything GEMM-shaped runs on WMMA.
// ---------------------------------------------------------------------------

#define DM   512
#define NH   8
#define HD   64
#define TB   2
#define TT   2048
#define ROWS (TB * TT)   // 4096
#define BH   (TB * NH)   // 16
#define CH   16          // chunk length
#define NC   (TT / CH)   // 128
#define EPSV 1e-6f

typedef __attribute__((ext_vector_type(2)))  float  v2f;
typedef __attribute__((ext_vector_type(8)))  float  v8f;
typedef __attribute__((ext_vector_type(16))) __bf16 v16bf;

union BfVec { unsigned u[8]; v16bf v; };

static __device__ __forceinline__ v8f wmma_f32x4(v2f a, v2f b, v8f c) {
  // V_WMMA_F32_16X16X4_F32  (exact fp32 matrix op for the recurrent state)
  return __builtin_amdgcn_wmma_f32_16x16x4_f32(false, a, false, b, (short)0, c,
                                               false, false);
}
static __device__ __forceinline__ v8f wmma_bf16(v16bf a, v16bf b, v8f c) {
  // V_WMMA_F32_16X16X32_BF16 (fast path for the big projections)
  return __builtin_amdgcn_wmma_f32_16x16x32_bf16(false, a, false, b, (short)0,
                                                 c, false, false);
}

static __device__ __forceinline__ unsigned short f2bf(float f) {
  unsigned u = __builtin_bit_cast(unsigned, f);
  unsigned r = u + 0x7FFFu + ((u >> 16) & 1u);   // round-to-nearest-even
  return (unsigned short)(r >> 16);
}

// ---------------------------------------------------------------------------
// Prep kernels (elementwise, memory-trivial)
// ---------------------------------------------------------------------------

__global__ void cvt_x_bf16(const float* __restrict__ x,
                           unsigned* __restrict__ xbf, int n2) {
  int i = blockIdx.x * blockDim.x + threadIdx.x;
  if (i >= n2) return;
  unsigned lo = f2bf(x[2 * i]);
  unsigned hi = f2bf(x[2 * i + 1]);
  xbf[i] = lo | (hi << 16);
}

// Pack W[2k][n], W[2k+1][n] into one dword: exactly the bf16 B-operand layout.
__global__ void pack_w(const float* __restrict__ W0, const float* __restrict__ W1,
                       const float* __restrict__ W2, const float* __restrict__ W3,
                       unsigned* __restrict__ P, int total) {
  int i = blockIdx.x * blockDim.x + threadIdx.x;
  if (i >= total) return;
  int w   = i >> 17;            // 256*512 = 131072 per weight
  int rem = i & 131071;
  int kp  = rem >> 9;
  int n   = rem & 511;
  const float* W = (w == 0) ? W0 : (w == 1) ? W1 : (w == 2) ? W2 : W3;
  unsigned lo = f2bf(W[(size_t)(2 * kp) * DM + n]);
  unsigned hi = f2bf(W[(size_t)(2 * kp + 1) * DM + n]);
  P[i] = lo | (hi << 16);
}

// ---------------------------------------------------------------------------
// bf16 WMMA GEMM tile: one wave computes a 16x16 tile of (X @ W), K = 512.
// A layout: lane holds row m=lane%16 with K pairs; B from packed weights.
// ---------------------------------------------------------------------------
static __device__ __forceinline__ v8f gemm_tile_bf16(
    const unsigned* __restrict__ xu,   // bf16-pair rows [ROWS][256]
    const unsigned* __restrict__ wu,   // packed W       [256][512]
    int row0, int ncol, int lane) {
  const int m    = lane & 15;
  const int hi   = lane >> 4;
  const int aoff = hi ? 8 : 0;   // lanes 16-31 hold K=16..31
  const int boff = hi ? 8 : 0;   // lanes 16-31 hold K=16..31
  v8f acc = {};
  const unsigned* xr = xu + (size_t)(row0 + m) * (DM / 2);
  for (int k0 = 0; k0 < DM; k0 += 32) {
    BfVec A, Bv;
    const unsigned* xk = xr + (k0 >> 1);
#pragma unroll
    for (int j = 0; j < 4; j++) {
      A.u[j]     = xk[aoff + j];       // K = base + 2j, 2j+1
      A.u[4 + j] = xk[4 + aoff + j];   // K = base + 8 + 2j, ...
    }
#pragma unroll
    for (int r = 0; r < 8; r++)
      Bv.u[r] = wu[(size_t)((k0 >> 1) + boff + r) * DM + ncol];
    acc = wmma_bf16(A.v, Bv.v, acc);
  }
  return acc;
}

// ---------------------------------------------------------------------------
// Fused QKV projection + bias + phi (ELU+1 on Q,K), write (B,H,T,D) fp32.
// 24576 waves total, exact grid (EXEC all-ones for WMMA).
// ---------------------------------------------------------------------------
__global__ void qkv_proj(const unsigned* __restrict__ xbf,
                         const unsigned* __restrict__ Wpk,  // q,k,v blocks
                         const float* __restrict__ bq, const float* __restrict__ bk,
                         const float* __restrict__ bv,
                         float* __restrict__ Q, float* __restrict__ K,
                         float* __restrict__ V) {
  const int lane = threadIdx.x & 31;
  const int wid  = blockIdx.x * 8 + (threadIdx.x >> 5);
  const int ctA  = wid % 96;     // 3 matrices * 32 col tiles
  const int rowtile = wid / 96;
  const int which = ctA >> 5;    // 0=Q 1=K 2=V
  const int ct    = ctA & 31;
  const int row0  = rowtile * 16;
  const int ncol  = ct * 16 + (lane & 15);

  const unsigned* wu = Wpk + (size_t)which * (DM / 2) * DM;
  v8f acc = gemm_tile_bf16(xbf, wu, row0, ncol, lane);

  const float* bias = (which == 0) ? bq : (which == 1) ? bk : bv;
  float* dst        = (which == 0) ? Q : (which == 1) ? K : V;
  const float bval  = bias[ncol];
  const int hi = lane >> 4;
  const int h = ncol >> 6, d = ncol & 63;
#pragma unroll
  for (int r = 0; r < 8; r++) {
    int row = row0 + r + 8 * hi;           // C layout: VGPR r -> M = r + 8*hi
    float v = acc[r] + bval;
    if (which < 2) v = (v > 0.0f) ? (v + 1.0f) : __expf(v);  // elu(x)+1
    int bb = row >> 11, t = row & 2047;
    dst[(((size_t)(bb * NH + h)) * TT + t) * HD + d] = v;
  }
}

// ---------------------------------------------------------------------------
// Pass 1: per-chunk M_c = K_chunk^T @ V_chunk (64x64, fp32 WMMA) + z_c sums.
// One block (4 waves) per (bh, chunk); wave rt owns 16 rows of M.
// ---------------------------------------------------------------------------
__global__ void chunk_kv(const float* __restrict__ K, const float* __restrict__ V,
                         float* __restrict__ Mc, float* __restrict__ zc) {
  const int bh   = blockIdx.x >> 7;
  const int c    = blockIdx.x & (NC - 1);
  const int lane = threadIdx.x & 31;
  const int rt   = threadIdx.x >> 5;   // 0..3
  const int t0   = c * CH;
  const float* Kb = K + (size_t)bh * TT * HD;
  const float* Vb = V + (size_t)bh * TT * HD;
  const int m = lane & 15, hi = lane >> 4, k2 = 2 * hi, nl = lane & 15;
  const int dA = rt * 16 + m;          // A = K^T row = feature d
  float* Mout = Mc + ((size_t)bh * NC + c) * (HD * HD);

#pragma unroll
  for (int n0 = 0; n0 < HD; n0 += 16) {
    v8f acc = {};
#pragma unroll
    for (int kb = 0; kb < CH; kb += 4) {
      v2f a; a.x = Kb[(size_t)(t0 + kb + k2) * HD + dA];
             a.y = Kb[(size_t)(t0 + kb + k2 + 1) * HD + dA];
      v2f b; b.x = Vb[(size_t)(t0 + kb + k2) * HD + n0 + nl];
             b.y = Vb[(size_t)(t0 + kb + k2 + 1) * HD + n0 + nl];
      acc = wmma_f32x4(a, b, acc);
    }
#pragma unroll
    for (int r = 0; r < 8; r++) {
      int dr = rt * 16 + r + 8 * hi;
      Mout[(size_t)dr * HD + n0 + nl] = acc[r];
    }
  }
  if (threadIdx.x < HD) {
    int d = threadIdx.x;
    float s = 0.0f;
    for (int t = 0; t < CH; t++) s += Kb[(size_t)(t0 + t) * HD + d];
    zc[((size_t)bh * NC + c) * HD + d] = s;
  }
}

// ---------------------------------------------------------------------------
// Pass 2: in-place exclusive prefix scan over the 128 chunks (per element).
// 16 bh * (4096 + 64) elements; stays resident in L2 (192 MB).
// ---------------------------------------------------------------------------
__global__ void scan_excl(float* __restrict__ Mc, float* __restrict__ zc) {
  int i = blockIdx.x * blockDim.x + threadIdx.x;
  const int per = HD * HD + HD;        // 4160
  if (i >= BH * per) return;
  int bh = i / per, e = i % per;
  if (e < HD * HD) {
    float acc = 0.0f;
    float* p = Mc + (size_t)bh * NC * HD * HD + e;
    for (int c = 0; c < NC; c++) { float t = p[(size_t)c * HD * HD]; p[(size_t)c * HD * HD] = acc; acc += t; }
  } else {
    int d = e - HD * HD;
    float acc = 0.0f;
    float* p = zc + (size_t)bh * NC * HD + d;
    for (int c = 0; c < NC; c++) { float t = p[(size_t)c * HD]; p[(size_t)c * HD] = acc; acc += t; }
  }
}

// ---------------------------------------------------------------------------
// Pass 3: out_chunk = Q@S_pre + mask(QK^T)@V, den = Q.z_pre + rowsum(mask P).
// One wave per (bh, chunk, 16-wide D tile).  52 fp32 WMMAs per wave.
// ---------------------------------------------------------------------------
__global__ void attn_chunk_out(const float* __restrict__ Q,
                               const float* __restrict__ K,
                               const float* __restrict__ V,
                               const float* __restrict__ Spre,
                               const float* __restrict__ zpre,
                               unsigned short* __restrict__ Obf) {
  __shared__ float Plds[8][16][17];
  const int lane = threadIdx.x & 31;
  const int ws   = threadIdx.x >> 5;
  const int unit = blockIdx.x * 8 + ws;
  const int ct = unit & 3;
  const int c  = (unit >> 2) & (NC - 1);
  const int bh = unit >> 9;
  const int dn = ct * 16;
  const int t0 = c * CH;

  const float* Qb = Q + (size_t)bh * TT * HD;
  const float* Kb = K + (size_t)bh * TT * HD;
  const float* Vb = V + (size_t)bh * TT * HD;
  const float* Sp = Spre + ((size_t)bh * NC + c) * (HD * HD);
  const float* zp = zpre + ((size_t)bh * NC + c) * HD;

  const int m = lane & 15, hi = lane >> 4, k2 = 2 * hi, nl = lane & 15;
  const float* Qrow = Qb + (size_t)(t0 + m) * HD;

  // Hoist Q A-operand fragments (shared by P, Pz and inter products).
  v2f aq[16];
#pragma unroll
  for (int kb = 0; kb < HD; kb += 4) {
    aq[kb >> 2].x = Qrow[kb + k2];
    aq[kb >> 2].y = Qrow[kb + k2 + 1];
  }

  // P = Q K_chunk^T  (16x16), Pz = Q z_pre (broadcast column).
  v8f p = {}, pz = {};
#pragma unroll
  for (int kb = 0; kb < HD; kb += 4) {
    v2f bk; bk.x = Kb[(size_t)(t0 + nl) * HD + kb + k2];
            bk.y = Kb[(size_t)(t0 + nl) * HD + kb + k2 + 1];
    v2f bz; bz.x = zp[kb + k2]; bz.y = zp[kb + k2 + 1];
    p  = wmma_f32x4(aq[kb >> 2], bk, p);
    pz = wmma_f32x4(aq[kb >> 2], bz, pz);
  }

  // Causal mask, stage P to LDS (A-operand needs a row-major redistribution),
  // denominator via butterfly row-sums over the 16-lane row groups.
  float den[8];
#pragma unroll
  for (int r = 0; r < 8; r++) {
    int mr = r + 8 * hi;
    float v = p[r];
    if (nl > mr) v = 0.0f;                 // keep t' <= t (incl. diagonal)
    Plds[ws][mr][nl] = v;
    float s = v;
    s += __shfl_xor(s, 1, 32);
    s += __shfl_xor(s, 2, 32);
    s += __shfl_xor(s, 4, 32);
    s += __shfl_xor(s, 8, 32);
    den[r] = fmaxf(s + pz[r], EPSV);
  }
  asm volatile("s_wait_dscnt 0" ::: "memory");

  // Inter: Q @ S_pre;  Intra: mask(P) @ V_chunk.
  v8f acc = {};
#pragma unroll
  for (int kb = 0; kb < HD; kb += 4) {
    v2f b; b.x = Sp[(size_t)(kb + k2) * HD + dn + nl];
           b.y = Sp[(size_t)(kb + k2 + 1) * HD + dn + nl];
    acc = wmma_f32x4(aq[kb >> 2], b, acc);
  }
#pragma unroll
  for (int kb = 0; kb < CH; kb += 4) {
    v2f a; a.x = Plds[ws][m][kb + k2]; a.y = Plds[ws][m][kb + k2 + 1];
    v2f b; b.x = Vb[(size_t)(t0 + kb + k2) * HD + dn + nl];
           b.y = Vb[(size_t)(t0 + kb + k2 + 1) * HD + dn + nl];
    acc = wmma_f32x4(a, b, acc);
  }

  // Normalize + regather heads to (B,T,d_model) in bf16 for the final GEMM.
  const int bb = bh >> 3, h = bh & 7;
#pragma unroll
  for (int r = 0; r < 8; r++) {
    int mr = r + 8 * hi;
    float val = acc[r] / den[r];
    int orow = bb * TT + t0 + mr;
    int ocol = h * HD + dn + nl;
    Obf[(size_t)orow * DM + ocol] = f2bf(val);
  }
}

// ---------------------------------------------------------------------------
// Output projection: out = O @ Wo + bo  (fp32 result to d_out).
// ---------------------------------------------------------------------------
__global__ void out_proj(const unsigned* __restrict__ obf,
                         const unsigned* __restrict__ Wo,
                         const float* __restrict__ bo,
                         float* __restrict__ out) {
  const int lane = threadIdx.x & 31;
  const int wid  = blockIdx.x * 8 + (threadIdx.x >> 5);
  const int ct   = wid & 31;
  const int rowtile = wid >> 5;
  const int row0 = rowtile * 16;
  const int ncol = ct * 16 + (lane & 15);
  v8f acc = gemm_tile_bf16(obf, Wo, row0, ncol, lane);
  const float bval = bo[ncol];
  const int hi = lane >> 4;
#pragma unroll
  for (int r = 0; r < 8; r++) {
    int row = row0 + r + 8 * hi;
    out[(size_t)row * DM + ncol] = acc[r] + bval;
  }
}

// ---------------------------------------------------------------------------
// Host launch
// ---------------------------------------------------------------------------
extern "C" void kernel_launch(void* const* d_in, const int* in_sizes, int n_in,
                              void* d_out, int out_size, void* d_ws,
                              size_t ws_size, hipStream_t stream) {
  const float* x  = (const float*)d_in[0];
  const float* Wq = (const float*)d_in[1];
  const float* bq = (const float*)d_in[2];
  const float* Wk = (const float*)d_in[3];
  const float* bk = (const float*)d_in[4];
  const float* Wv = (const float*)d_in[5];
  const float* bv = (const float*)d_in[6];
  const float* Wo = (const float*)d_in[7];
  const float* bo = (const float*)d_in[8];
  float* out = (float*)d_out;

  char* ws = (char*)d_ws;
  unsigned* xbf = (unsigned*)ws;            ws += (size_t)ROWS * (DM / 2) * 4;  // 4 MB
  unsigned* Wpk = (unsigned*)ws;            ws += (size_t)4 * (DM / 2) * DM * 4; // 2 MB
  float* Q      = (float*)ws;               ws += (size_t)BH * TT * HD * 4;     // 8 MB
  float* K      = (float*)ws;               ws += (size_t)BH * TT * HD * 4;     // 8 MB
  float* V      = (float*)ws;               ws += (size_t)BH * TT * HD * 4;     // 8 MB
  float* Mc     = (float*)ws;               ws += (size_t)BH * NC * HD * HD * 4; // 32 MB
  float* zc     = (float*)ws;               ws += (size_t)BH * NC * HD * 4;     // 0.5 MB
  unsigned short* Obf = (unsigned short*)ws; ws += (size_t)ROWS * DM * 2;       // 4 MB

  cvt_x_bf16<<<(ROWS * DM / 2 + 255) / 256, 256, 0, stream>>>(x, xbf,
                                                              ROWS * DM / 2);
  pack_w<<<(4 * 131072 + 255) / 256, 256, 0, stream>>>(Wq, Wk, Wv, Wo, Wpk,
                                                       4 * 131072);
  qkv_proj<<<3072, 256, 0, stream>>>(xbf, Wpk, bq, bk, bv, Q, K, V);
  chunk_kv<<<BH * NC, 128, 0, stream>>>(K, V, Mc, zc);
  scan_excl<<<(BH * (HD * HD + HD) + 255) / 256, 256, 0, stream>>>(Mc, zc);
  attn_chunk_out<<<1024, 256, 0, stream>>>(Q, K, V, Mc, zc, Obf);
  out_proj<<<1024, 256, 0, stream>>>((const unsigned*)Obf, Wpk + 3 * 131072,
                                     bo, out);
}
